// BigBirdAttention_3040836845888
// MI455X (gfx1250) — compile-verified, matrix operand order checked by reading, not run on hardware
//
#include <hip/hip_runtime.h>
#include <hip/hip_bf16.h>

// ---------------- problem constants ----------------
#define BB    2
#define SS    4096
#define HID   768
#define NHEAD 12
#define HDIM  64
#define BSZ   64
#define NBLK  64
#define NEGV  (-1000000000.0f)

typedef __bf16 bf16;
typedef __attribute__((ext_vector_type(16))) bf16  v16bf;
typedef __attribute__((ext_vector_type(8)))  bf16  v8bf;
typedef __attribute__((ext_vector_type(8)))  float v8f;
typedef __attribute__((ext_vector_type(4)))  float v4f;
typedef __attribute__((ext_vector_type(4)))  int   v4i;
typedef __attribute__((ext_vector_type(4)))  unsigned int v4u;

#ifndef __has_builtin
#define __has_builtin(x) 0
#endif

#if __has_builtin(__builtin_amdgcn_global_load_async_to_lds_b128) && \
    __has_builtin(__builtin_amdgcn_s_wait_asynccnt)
#define USE_ASYNC 1
#else
#define USE_ASYNC 0
#endif

// 16-byte global -> LDS copy; async (ASYNCcnt-tracked) when available.
// Builtin prototype (from compiler diagnostic): arg0 = v4i in AS(1) [__device__],
// arg1 = v4i in AS(3) [__shared__], then two integer immediates (offset, cpol).
__device__ __forceinline__ void cp16_g2l(const bf16* g, bf16* l) {
#if USE_ASYNC
    v4i* gv = (v4i*)(void*)g;   // retype (generic)
    v4i* lv = (v4i*)(void*)l;   // retype (generic)
    __builtin_amdgcn_global_load_async_to_lds_b128(
        (__attribute__((address_space(1))) v4i*)gv,
        (__attribute__((address_space(3))) v4i*)lv,
        0, 0);
#else
    *(v4u*)l = *(const v4u*)g;
#endif
}

__device__ __forceinline__ void wait_async_done() {
#if USE_ASYNC
    __builtin_amdgcn_s_wait_asynccnt(0);
#endif
}

// ---------------- WMMA helpers ----------------
__device__ __forceinline__ v8f wmma_bf16(v16bf a, v16bf b, v8f c) {
    // D = A(16x32 bf16) * B(32x16 bf16) + C(16x16 f32)
    return __builtin_amdgcn_wmma_f32_16x16x32_bf16(
        /*neg_a*/false, a, /*neg_b*/false, b,
        /*c_mod*/(short)0, c, /*reuse_a*/false, /*reuse_b*/false);
}

// Fragment load from row-major bf16 source (global or LDS). Lane holds row
// (lane&15); k-elements {koff+0..7, 16+koff+0..7}, koff = (lane>>4)*8.
__device__ __forceinline__ v16bf ld_frag_bf16(const bf16* base, int ld) {
    int lane = threadIdx.x & 31;
    int r    = lane & 15;
    int koff = (lane >> 4) << 3;
    const bf16* p = base + (size_t)r * ld + koff;
    v8bf lo = *(const v8bf*)(p);
    v8bf hi = *(const v8bf*)(p + 16);
    v16bf f;
#pragma unroll
    for (int i = 0; i < 8; ++i) { f[i] = lo[i]; f[i + 8] = hi[i]; }
    return f;
}

// Same, but source is f32 (converted on the fly to bf16).
__device__ __forceinline__ v16bf ld_frag_f32(const float* base, int ld) {
    int lane = threadIdx.x & 31;
    int r    = lane & 15;
    int koff = (lane >> 4) << 3;
    const float* p = base + (size_t)r * ld + koff;
    v4f a = *(const v4f*)(p);
    v4f b = *(const v4f*)(p + 4);
    v4f c = *(const v4f*)(p + 16);
    v4f d = *(const v4f*)(p + 20);
    v16bf f;
#pragma unroll
    for (int i = 0; i < 4; ++i) {
        f[i]      = (bf16)a[i];
        f[i + 4]  = (bf16)b[i];
        f[i + 8]  = (bf16)c[i];
        f[i + 12] = (bf16)d[i];
    }
    return f;
}

// ---------------- weight f32 -> bf16 conversion ----------------
__global__ void cvt_f32_bf16(const float* __restrict__ src, bf16* __restrict__ dst, int n) {
    int i = blockIdx.x * blockDim.x + threadIdx.x;
    if (i < n) dst[i] = (bf16)src[i];
}

// ---------------- key-block index table ----------------
__global__ void build_idx(const int* __restrict__ rand_idx, int* __restrict__ tab) {
    int qb = blockIdx.x;
    if (threadIdx.x == 0) {
        tab[qb * 8 + 0] = (qb + NBLK - 1) & (NBLK - 1);
        tab[qb * 8 + 1] = qb;
        tab[qb * 8 + 2] = (qb + 1) & (NBLK - 1);
        tab[qb * 8 + 3] = 0;
        tab[qb * 8 + 4] = NBLK - 1;
        tab[qb * 8 + 5] = rand_idx[qb * 3 + 0];
        tab[qb * 8 + 6] = rand_idx[qb * 3 + 1];
        tab[qb * 8 + 7] = rand_idx[qb * 3 + 2];
    }
}

// ---------------- QKV projection GEMM (WMMA) ----------------
// mode: 0 = Q (scaled 1/8, [B,NH,S,64] bf16), 1 = K (same), 2 = V^T ([B,NH,64,S] bf16)
__global__ __launch_bounds__(128) void proj_qkv(
    const float* __restrict__ X, const bf16* __restrict__ W16,
    const float* __restrict__ bias, bf16* __restrict__ out, int mode) {
    int w    = threadIdx.x >> 5;
    int lane = threadIdx.x & 31;
    int h    = blockIdx.y;
    int row0 = blockIdx.x * 64 + w * 16;

    v8f acc[4];
#pragma unroll
    for (int j = 0; j < 4; ++j) acc[j] = {};

    for (int k0 = 0; k0 < HID; k0 += 32) {
        v16bf af = ld_frag_f32(X + (size_t)row0 * HID + k0, HID);
#pragma unroll
        for (int j = 0; j < 4; ++j) {
            v16bf bfg = ld_frag_bf16(W16 + (size_t)(h * 64 + j * 16) * HID + k0, HID);
            acc[j] = wmma_bf16(af, bfg, acc[j]);
        }
    }

    int koff8 = (lane >> 4) << 3;
    float scl = (mode == 0) ? 0.125f : 1.0f;  // 1/sqrt(HD)
#pragma unroll
    for (int j = 0; j < 4; ++j) {
        int d = j * 16 + (lane & 15);
        float bj = bias[h * 64 + d];
#pragma unroll
        for (int r = 0; r < 8; ++r) {
            float v = (acc[j][r] + bj) * scl;
            int row = row0 + r + koff8;
            int b   = row >> 12;
            int s   = row & (SS - 1);
            size_t bh = (size_t)b * NHEAD + h;
            if (mode <= 1)
                out[(bh * SS + s) * HDIM + d] = (bf16)v;
            else
                out[(bh * HDIM + d) * SS + s] = (bf16)v;
        }
    }
}

// ---------------- BigBird attention (flash-style, WMMA, async LDS staging) -----
// grid = (NB, NH, B), 128 threads = 4 waves; wave w handles 16 queries.
// Each 64-key chunk's K (8KB, contiguous) and V^T rows (8KB) are async-copied to
// LDS double buffers; copy of chunk kc+1 overlaps WMMA compute of chunk kc.
// Scores computed transposed (K @ Q^T) so the f32 D-fragment reinterprets directly
// as the bf16 A-fragment of P for the P @ V WMMA.
__global__ __launch_bounds__(128) void bigbird_attn(
    const bf16* __restrict__ Q, const bf16* __restrict__ K, const bf16* __restrict__ Vt,
    const int* __restrict__ tab, const float* __restrict__ bmask,
    const float* __restrict__ to_mask, const float* __restrict__ from_mask,
    bf16* __restrict__ ctx) {
    int qb = blockIdx.x, h = blockIdx.y, b = blockIdx.z;
    int tid = threadIdx.x;
    int w = tid >> 5, lane = tid & 31;
    int bh = b * NHEAD + h;
    bool is_glob = (qb == 0) || (qb == NBLK - 1);
    int cnt = is_glob ? NBLK : 8;
    int koff8 = (lane >> 4) << 3;

    __shared__ __align__(16) bf16 skbuf[2][BSZ * HDIM];   // K block [key][hd]
    __shared__ __align__(16) bf16 svbuf[2][HDIM * BSZ];   // V^T block [hd][key]
    __shared__ float sbias[2][BSZ];

    const bf16* Kg  = K  + (size_t)bh * SS * HDIM;
    const bf16* Vtg = Vt + (size_t)bh * HDIM * SS;

    // Q^T B-fragments for this wave's 16 queries (loaded once)
    const bf16* Qb = Q + ((size_t)bh * SS + qb * 64 + w * 16) * HDIM;
    v16bf qf0 = ld_frag_bf16(Qb, HDIM);        // hd 0..31
    v16bf qf1 = ld_frag_bf16(Qb + 32, HDIM);   // hd 32..63

    float m = -1e30f, l = 0.0f;
    v8f acc[4];
#pragma unroll
    for (int j = 0; j < 4; ++j) acc[j] = {};

    auto kb_of = [&](int kc) { return is_glob ? kc : tab[qb * 8 + kc]; };
    auto stage = [&](int kc, int buf) {
        int kb = kb_of(kc);
        // K tile: 64 rows x 64 bf16, contiguous 8KB
        const bf16* kg = Kg + (size_t)kb * 64 * HDIM;
#pragma unroll
        for (int c = 0; c < 4; ++c) {
            int idx = tid + c * 128;           // 512 x 16B chunks
            cp16_g2l(kg + idx * 8, &skbuf[buf][idx * 8]);
        }
        // V^T tile: 64 rows (hd) x 64 keys, row stride S
#pragma unroll
        for (int c = 0; c < 4; ++c) {
            int idx = tid + c * 128;
            int row = idx >> 3, c8 = (idx & 7) * 8;
            cp16_g2l(Vtg + (size_t)row * SS + kb * 64 + c8, &svbuf[buf][row * 64 + c8]);
        }
        // additive mask bias
        if (tid < BSZ) {
            float mk = is_glob ? to_mask[(size_t)b * SS + kb * 64 + tid]
                               : bmask[((size_t)b * NBLK + kb) * BSZ + tid];
            sbias[buf][tid] = (1.0f - mk) * NEGV;
        }
    };

    stage(0, 0);

    for (int kc = 0; kc < cnt; ++kc) {
        int cur = kc & 1;
        wait_async_done();       // this wave's copies for chunk kc have landed
        __syncthreads();         // all waves' copies visible; compute(kc-1) done
        if (kc + 1 < cnt) stage(kc + 1, cur ^ 1);

        const bf16* skb = skbuf[cur];
        const bf16* svb = svbuf[cur];

        // S^T tiles from LDS: keys in VGPRs, queries in lanes
        v8f s[4];
#pragma unroll
        for (int t = 0; t < 4; ++t) {
            v16bf kf0 = ld_frag_bf16(skb + t * 16 * HDIM, HDIM);
            v16bf kf1 = ld_frag_bf16(skb + t * 16 * HDIM + 32, HDIM);
            v8f z = {};
            s[t] = wmma_bf16(kf0, qf0, z);
            s[t] = wmma_bf16(kf1, qf1, s[t]);
        }

        // mask bias + chunk max (per query = lane&15, combine halves)
        float cm = -1e30f;
#pragma unroll
        for (int t = 0; t < 4; ++t)
#pragma unroll
            for (int r = 0; r < 8; ++r) {
                float v = s[t][r] + sbias[cur][t * 16 + r + koff8];
                s[t][r] = v;
                cm = fmaxf(cm, v);
            }
        cm = fmaxf(cm, __shfl_xor(cm, 16));
        float mn   = fmaxf(m, cm);
        float resc = __expf(m - mn);
        float lsum = 0.0f;
#pragma unroll
        for (int t = 0; t < 4; ++t)
#pragma unroll
            for (int r = 0; r < 8; ++r) {
                float p = __expf(s[t][r] - mn);
                s[t][r] = p;
                lsum += p;
            }
        lsum += __shfl_xor(lsum, 16);
        l = l * resc + lsum;
        m = mn;

        // rescale ctx accumulators (row of acc is query r + koff8)
        float rf[8];
#pragma unroll
        for (int r = 0; r < 8; ++r) rf[r] = __shfl(resc, r + koff8);
#pragma unroll
        for (int j = 0; j < 4; ++j)
#pragma unroll
            for (int r = 0; r < 8; ++r) acc[j][r] *= rf[r];

        // P A-fragments straight from the S^T D-fragments (layout identity)
        v16bf pa0, pa1;
#pragma unroll
        for (int i = 0; i < 8; ++i) {
            pa0[i] = (bf16)s[0][i];  pa0[i + 8] = (bf16)s[1][i];
            pa1[i] = (bf16)s[2][i];  pa1[i + 8] = (bf16)s[3][i];
        }

        // ctx += P @ V   (V^T rows are hd, contiguous over keys, from LDS)
#pragma unroll
        for (int j = 0; j < 4; ++j) {
            v16bf vb0 = ld_frag_bf16(svb + j * 16 * HDIM, HDIM);        // keys 0..31
            v16bf vb1 = ld_frag_bf16(svb + j * 16 * HDIM + 32, HDIM);   // keys 32..63
            acc[j] = wmma_bf16(pa0, vb0, acc[j]);
            acc[j] = wmma_bf16(pa1, vb1, acc[j]);
        }
    }

    // finalize: divide by l, apply from_mask, store bf16 ctx [B,S,NH*HD]
    float invl = 1.0f / l;
    float fm   = from_mask[(size_t)b * SS + qb * 64 + w * 16 + (lane & 15)];
    float wgt  = invl * fm;
    float wr[8];
#pragma unroll
    for (int r = 0; r < 8; ++r) wr[r] = __shfl(wgt, r + koff8);
#pragma unroll
    for (int j = 0; j < 4; ++j) {
        int col = h * 64 + j * 16 + (lane & 15);
#pragma unroll
        for (int r = 0; r < 8; ++r) {
            int srow = qb * 64 + w * 16 + r + koff8;
            ctx[((size_t)b * SS + srow) * HID + col] = (bf16)(acc[j][r] * wr[r]);
        }
    }
}

// ---------------- output projection GEMM (WMMA) ----------------
__global__ __launch_bounds__(128) void proj_out(
    const bf16* __restrict__ ctx, const bf16* __restrict__ Wo16,
    const float* __restrict__ bias, float* __restrict__ out) {
    int w    = threadIdx.x >> 5;
    int lane = threadIdx.x & 31;
    int h    = blockIdx.y;
    int row0 = blockIdx.x * 64 + w * 16;

    v8f acc[4];
#pragma unroll
    for (int j = 0; j < 4; ++j) acc[j] = {};

    for (int k0 = 0; k0 < HID; k0 += 32) {
        v16bf af = ld_frag_bf16(ctx + (size_t)row0 * HID + k0, HID);
#pragma unroll
        for (int j = 0; j < 4; ++j) {
            v16bf bfg = ld_frag_bf16(Wo16 + (size_t)(h * 64 + j * 16) * HID + k0, HID);
            acc[j] = wmma_bf16(af, bfg, acc[j]);
        }
    }

    int koff8 = (lane >> 4) << 3;
#pragma unroll
    for (int j = 0; j < 4; ++j) {
        int d = h * 64 + j * 16 + (lane & 15);
        float bj = bias[d];
#pragma unroll
        for (int r = 0; r < 8; ++r) {
            int row = row0 + r + koff8;
            out[(size_t)row * HID + d] = acc[j][r] + bj;
        }
    }
}

// ---------------- host launch ----------------
extern "C" void kernel_launch(void* const* d_in, const int* in_sizes, int n_in,
                              void* d_out, int out_size, void* d_ws, size_t ws_size,
                              hipStream_t stream) {
    const float* X        = (const float*)d_in[0];
    // d_in[1] = band_mask (unused by reference)
    const float* from_m   = (const float*)d_in[2];
    const float* to_m     = (const float*)d_in[3];
    const float* bmask    = (const float*)d_in[4];
    const int*   rand_idx = (const int*)d_in[5];
    const float* Wq_w = (const float*)d_in[6];
    const float* Wq_b = (const float*)d_in[7];
    const float* Wk_w = (const float*)d_in[8];
    const float* Wk_b = (const float*)d_in[9];
    const float* Wv_w = (const float*)d_in[10];
    const float* Wv_b = (const float*)d_in[11];
    const float* Wo_w = (const float*)d_in[12];
    const float* Wo_b = (const float*)d_in[13];
    float* out = (float*)d_out;

    char* ws = (char*)d_ws;
    const size_t QELEM = (size_t)BB * NHEAD * SS * HDIM;
    const size_t WELEM = (size_t)HID * HID;
    size_t off = 0;
    bf16* Qbuf  = (bf16*)(ws + off); off += QELEM * 2;
    bf16* Kbuf  = (bf16*)(ws + off); off += QELEM * 2;
    bf16* Vtbuf = (bf16*)(ws + off); off += QELEM * 2;
    bf16* ctx   = (bf16*)(ws + off); off += (size_t)BB * SS * HID * 2;
    bf16* Wq16  = (bf16*)(ws + off); off += WELEM * 2;
    bf16* Wk16  = (bf16*)(ws + off); off += WELEM * 2;
    bf16* Wv16  = (bf16*)(ws + off); off += WELEM * 2;
    bf16* Wo16  = (bf16*)(ws + off); off += WELEM * 2;
    int*  tab   = (int*)(ws + off);  off += (size_t)NBLK * 8 * 4;

    int ncvt = (int)WELEM;
    int gcvt = (ncvt + 255) / 256;
    cvt_f32_bf16<<<gcvt, 256, 0, stream>>>(Wq_w, Wq16, ncvt);
    cvt_f32_bf16<<<gcvt, 256, 0, stream>>>(Wk_w, Wk16, ncvt);
    cvt_f32_bf16<<<gcvt, 256, 0, stream>>>(Wv_w, Wv16, ncvt);
    cvt_f32_bf16<<<gcvt, 256, 0, stream>>>(Wo_w, Wo16, ncvt);

    build_idx<<<NBLK, 32, 0, stream>>>(rand_idx, tab);

    dim3 gproj((BB * SS) / 64, NHEAD);
    proj_qkv<<<gproj, 128, 0, stream>>>(X, Wq16, Wq_b, Qbuf, 0);
    proj_qkv<<<gproj, 128, 0, stream>>>(X, Wk16, Wk_b, Kbuf, 1);
    proj_qkv<<<gproj, 128, 0, stream>>>(X, Wv16, Wv_b, Vtbuf, 2);

    dim3 gattn(NBLK, NHEAD, BB);
    bigbird_attn<<<gattn, 128, 0, stream>>>(Qbuf, Kbuf, Vtbuf, tab, bmask, to_m, from_m, ctx);

    proj_out<<<gproj, 128, 0, stream>>>(ctx, Wo16, Wo_b, out);
}
